// GRU_10599979287102
// MI455X (gfx1250) — compile-verified, matrix operand order checked by reading, not run on hardware
//
#include <hip/hip_runtime.h>
#include <hip/hip_bf16.h>
#include <math.h>

#define B_  256
#define T_  1024
#define H_  256
#define P_  16
#define TH3 768   /* 3*H */

typedef __attribute__((ext_vector_type(16))) __bf16       v16bf;
typedef __attribute__((ext_vector_type(8)))  float        v8f;
typedef __attribute__((ext_vector_type(4)))  __bf16       v4bf;
typedef __attribute__((ext_vector_type(4)))  unsigned int v4u;
typedef __attribute__((ext_vector_type(8)))  int          v8i_t;
typedef __attribute__((ext_vector_type(4)))  int          v4i_t;

// ---------------------------------------------------------------------------
// Precision-conversion kernels (parallel, HBM-bandwidth bound, trivial)
// ---------------------------------------------------------------------------
__global__ __launch_bounds__(256) void cvt_x_kernel(const float* __restrict__ x,
                                                    __bf16* __restrict__ xb, long n4) {
  long i = (long)blockIdx.x * blockDim.x + threadIdx.x;
  if (i >= n4) return;
  const float4 v = ((const float4*)x)[i];
  v4bf o;
  o.x = (__bf16)v.x; o.y = (__bf16)v.y; o.z = (__bf16)v.z; o.w = (__bf16)v.w;
  ((v4bf*)xb)[i] = o;
}

__global__ __launch_bounds__(256) void cvt_w_kernel(const float* __restrict__ Wih,
                                                    const float* __restrict__ Whh,
                                                    __bf16* __restrict__ wih0,
                                                    __bf16* __restrict__ whh0,
                                                    __bf16* __restrict__ wih1,
                                                    __bf16* __restrict__ whh1) {
  const int M = TH3 * H_;
  int i   = blockIdx.x * 256 + threadIdx.x;   // 0 .. 4*M-1
  int m   = i / M;
  int off = i % M;
  const float* s;
  __bf16* d;
  switch (m) {
    case 0:  s = Wih + off;     d = wih0 + off; break;
    case 1:  s = Whh + off;     d = whh0 + off; break;
    case 2:  s = Wih + M + off; d = wih1 + off; break;
    default: s = Whh + M + off; d = whh1 + off; break;
  }
  *d = (__bf16)(*s);
}

// ---------------------------------------------------------------------------
// TDM: async load of one [16 x 256] bf16 tile (row stride T*H elems) into LDS
// D# group0/group1 per CDNA5 ISA sec 8.3/8.4; groups 2/3 unused (2D tensor).
// amdgpu-toolchain (clang-23) builtin arity: 6 args, extra int32x8 group.
// ---------------------------------------------------------------------------
__device__ __forceinline__ void tdm_load_xtile(const __bf16* gsrc, unsigned lds_off) {
  unsigned long long ga = (unsigned long long)(uintptr_t)gsrc;
  v4u g0 = { 1u,                                   // count = 1 (valid descriptor)
             lds_off,                              // lds_addr (bytes)
             (unsigned)(ga & 0xFFFFFFFFu),         // global_addr[31:0]
             (unsigned)((ga >> 32) & 0x01FFFFFFu)  // global_addr[56:32]
               | (2u << 30) };                     // type = 2 ("image")
  v8i_t g1 = { (int)(1u << 16),       // data_size = 1 (2-byte elements)
               (int)(256u << 16),     // tensor_dim0[15:0]  = 256  (bits 63:48)
               (int)(16u  << 16),     // tensor_dim1[15:0]  = 16   (bits 95:80)
               (int)(256u << 16),     // tile_dim0          = 256  (bits 127:112)
               (int)16,               // tile_dim1          = 16   (bits 143:128)
               (int)(T_ * H_),        // tensor_dim0_stride = 262144 (bits 191:160)
               0, 0 };
  v4i_t z4 = { 0, 0, 0, 0 };
  v8i_t z8 = { 0, 0, 0, 0, 0, 0, 0, 0 };
  __builtin_amdgcn_tensor_load_to_lds(g0, g1, z4, z4, z8, 0);
}

// ---------------------------------------------------------------------------
// One GEMM panel: C[16,768] (f32, LDS) = A[16,256] (bf16, LDS) * W^T
// W is [768,256] bf16 row-major in global (L2-resident).
// 48 N-tiles of 16 cols, 8 K-chunks of 32 -> v_wmma_f32_16x16x32_bf16.
// 16 waves; each wave owns 3 N-tiles; A fragments kept resident in VGPRs.
// ---------------------------------------------------------------------------
__device__ __forceinline__ void gemm_panel(const __bf16* A,
                                           const __bf16* __restrict__ W,
                                           float* C) {
  const int lane = threadIdx.x & 31;
  const int wave = threadIdx.x >> 5;
  const int arow = lane & 15;
  const int ksel = (lane >> 4) << 4;           // lanes 16-31 take upper K half

  v16bf a[8];
#pragma unroll
  for (int k = 0; k < 8; ++k)
    a[k] = *(const v16bf*)(A + arow * H_ + k * 32 + ksel);

  for (int nt = wave; nt < 48; nt += 16) {
    v8f acc = { 0.f, 0.f, 0.f, 0.f, 0.f, 0.f, 0.f, 0.f };
    const int col = nt * 16 + (lane & 15);
#pragma unroll
    for (int k = 0; k < 8; ++k) {
      v16bf b = *(const v16bf*)(W + (size_t)col * H_ + k * 32 + ksel);
      acc = __builtin_amdgcn_wmma_f32_16x16x32_bf16(
                false, a[k], false, b, (short)0, acc, false, false);
    }
    const int rb = (lane >> 4) * 8;             // C layout: lanes 0-15 M=0..7
#pragma unroll
    for (int r = 0; r < 8; ++r)
      C[(rb + r) * TH3 + col] = acc[r];
  }
}

// ---------------------------------------------------------------------------
// GRU gate math: h = (1-z)*n + z*h  with r,z,n from gi/gh (+biases)
// ---------------------------------------------------------------------------
__device__ __forceinline__ void gate_update(__bf16* h, const float* gi, const float* gh,
                                            const float* __restrict__ bih,
                                            const float* __restrict__ bhh) {
  for (int i = threadIdx.x; i < 16 * H_; i += 512) {
    const int row = i >> 8;
    const int col = i & (H_ - 1);
    const float* gir = gi + row * TH3;
    const float* ghr = gh + row * TH3;
    float r  = gir[col]          + bih[col]          + ghr[col]          + bhh[col];
    float z  = gir[H_ + col]     + bih[H_ + col]     + ghr[H_ + col]     + bhh[H_ + col];
    float in = gir[2 * H_ + col] + bih[2 * H_ + col];
    float hn = ghr[2 * H_ + col] + bhh[2 * H_ + col];
    r = 1.f / (1.f + __expf(-r));
    z = 1.f / (1.f + __expf(-z));
    float n  = tanhf(in + r * hn);
    float hp = (float)h[i];
    h[i] = (__bf16)((1.f - z) * n + z * hp);
  }
}

// ---------------------------------------------------------------------------
// Scan kernel: 16 blocks x 512 threads; block b owns batch rows [16b,16b+16)
// and runs the full T=1024 recurrence privately (batch rows are independent).
// ---------------------------------------------------------------------------
__global__ __launch_bounds__(512) void gru_scan_kernel(
    const __bf16* __restrict__ xb,
    const __bf16* __restrict__ wih0, const __bf16* __restrict__ whh0,
    const __bf16* __restrict__ wih1, const __bf16* __restrict__ whh1,
    const float* __restrict__ b_ih, const float* __restrict__ b_hh,
    const float* __restrict__ fc_w, const float* __restrict__ fc_b,
    float* __restrict__ out) {
  __shared__ __align__(32) __bf16 sh_xt[2][16 * H_];   // x_t tile, double buffered
  __shared__ __align__(32) __bf16 sh_h0[16 * H_];
  __shared__ __align__(32) __bf16 sh_h1[16 * H_];
  __shared__ float sh_gi[16 * TH3];
  __shared__ float sh_gh[16 * TH3];

  const int tid = threadIdx.x;
  const int wg  = blockIdx.x;
  const __bf16* xg = xb + (size_t)wg * 16 * T_ * H_;

  for (int i = tid; i < 16 * H_; i += 512) {
    sh_h0[i] = (__bf16)0.f;
    sh_h1[i] = (__bf16)0.f;
  }

  // prologue: async-load tile for t=0 (wave 0 drives the TDM)
  if (tid < 32)
    tdm_load_xtile(xg, (unsigned)(uintptr_t)&sh_xt[0][0]);

#pragma unroll 1
  for (int t = 0; t < T_; ++t) {
    const int cur = t & 1;
    if (tid < 32) {
      if (t + 1 < T_) {  // prefetch next tile, then wait only for the current one
        tdm_load_xtile(xg + (size_t)(t + 1) * H_,
                       (unsigned)(uintptr_t)&sh_xt[cur ^ 1][0]);
        __builtin_amdgcn_s_wait_tensorcnt(1);
      } else {
        __builtin_amdgcn_s_wait_tensorcnt(0);
      }
    }
    __syncthreads();

    // layer 0
    gemm_panel(&sh_xt[cur][0], wih0, sh_gi);
    gemm_panel(sh_h0,          whh0, sh_gh);
    __syncthreads();
    gate_update(sh_h0, sh_gi, sh_gh, b_ih, b_hh);
    __syncthreads();

    // layer 1 (input = new h0)
    gemm_panel(sh_h0, wih1, sh_gi);
    gemm_panel(sh_h1, whh1, sh_gh);
    __syncthreads();
    gate_update(sh_h1, sh_gi, sh_gh, b_ih + TH3, b_hh + TH3);
    __syncthreads();
  }

  // FC head: out[b,p] = h1[row,:] . fc_w[p,:] + fc_b[p]
  if (tid < 256) {
    const int row = tid >> 4;
    const int p   = tid & 15;
    float acc = fc_b[p];
    for (int j = 0; j < H_; ++j)
      acc += (float)sh_h1[row * H_ + j] * fc_w[p * H_ + j];
    out[(size_t)(wg * 16 + row) * P_ + p] = acc;
  }
}

// ---------------------------------------------------------------------------
extern "C" void kernel_launch(void* const* d_in, const int* in_sizes, int n_in,
                              void* d_out, int out_size, void* d_ws, size_t ws_size,
                              hipStream_t stream) {
  const float* x    = (const float*)d_in[0];
  const float* W_ih = (const float*)d_in[1];
  const float* W_hh = (const float*)d_in[2];
  const float* b_ih = (const float*)d_in[3];
  const float* b_hh = (const float*)d_in[4];
  const float* fc_w = (const float*)d_in[5];
  const float* fc_b = (const float*)d_in[6];
  float* out = (float*)d_out;

  // workspace: xb (bf16 x), then 4 bf16 weight matrices
  __bf16* xb = (__bf16*)d_ws;
  const size_t XB = (size_t)B_ * T_ * H_;
  const size_t WM = (size_t)TH3 * H_;
  __bf16* wih0 = xb + XB;
  __bf16* whh0 = wih0 + WM;
  __bf16* wih1 = whh0 + WM;
  __bf16* whh1 = wih1 + WM;

  const long n4 = (long)(XB / 4);
  cvt_x_kernel<<<(unsigned)(n4 / 256), 256, 0, stream>>>(x, xb, n4);
  cvt_w_kernel<<<(unsigned)((4 * WM) / 256), 256, 0, stream>>>(
      W_ih, W_hh, wih0, whh0, wih1, whh1);
  gru_scan_kernel<<<B_ / 16, 512, 0, stream>>>(
      xb, wih0, whh0, wih1, whh1, b_ih, b_hh, fc_w, fc_b, out);
}
